// Model_57183194578962
// MI455X (gfx1250) — matrix operand order repaired and checked by reading, not compiled, not run on hardware
//
#include <hip/hip_runtime.h>

// ---------------- model dims ----------------
#define B_      512
#define N_      100
#define E_      512
#define H_      8
#define DM_     1024     // 2*E
#define DEPTH_  128      // DM/H
#define NEG_INF_F (-1e9f)

typedef unsigned int   u32;
typedef unsigned short u16;
typedef __attribute__((ext_vector_type(4)))  u32    u32x4;
typedef __attribute__((ext_vector_type(16))) __bf16 v16bf;
typedef __attribute__((ext_vector_type(8)))  float  v8f;

union Frag { u32x4 q[2]; v16bf v; };

__device__ __forceinline__ u16 f2bf(float f) {
  u32 u = __float_as_uint(f);
  u32 r = (u + 0x7FFFu + ((u >> 16) & 1u)) >> 16;   // round-to-nearest-even
  return (u16)r;
}
__device__ __forceinline__ float bf2f(u16 h) {
  return __uint_as_float(((u32)h) << 16);
}

// ---------------------------------------------------------------------------
// Pack f32 row-major W[K,N] into WMMA B-fragment order (bf16):
// dword index = ((kt*ntiles + nt)*32 + lane)*8 + v
//   lane -> N = nt*16 + lane%16 ; koff = lane<16 ? 0 : 8
//   dword v -> {K = kt*32+koff+2(v&3)+16(v>>2) (lo16), K+1 (hi16)}
// ---------------------------------------------------------------------------
__global__ void __launch_bounds__(256)
pack_w(const float* __restrict__ W, u32* __restrict__ Wpk, int K, int N) {
  int idx    = blockIdx.x * 256 + threadIdx.x;   // one dword per thread
  int v      = idx & 7;
  int lane   = (idx >> 3) & 31;
  int tile   = idx >> 8;                          // kt*ntiles + nt
  int ntiles = N >> 4;
  int kt = tile / ntiles, nt = tile - kt * ntiles;
  int koff = (lane < 16) ? 0 : 8;
  int n = nt * 16 + (lane & 15);
  int k = kt * 32 + koff + (v & 3) * 2 + (v >> 2) * 16;
  u32 lo = f2bf(W[(size_t)k * N + n]);
  u32 hi = f2bf(W[(size_t)(k + 1) * N + n]);
  Wpk[idx] = lo | (hi << 16);
}

// ---------------------------------------------------------------------------
// Build x = concat(text_table[ids], neighbors_pos @ pos_W + pos_b) as bf16,
// plus padding mask vector m[b,n] = (id == 0).
// ---------------------------------------------------------------------------
__global__ void __launch_bounds__(256)
build_x(const int* __restrict__ ids,
        const float* __restrict__ npos,
        const float* __restrict__ text_table,
        const float* __restrict__ posW,
        const float* __restrict__ posb,
        u32* __restrict__ X,        // [B*N, 512] dwords (1024 bf16)
        float* __restrict__ mvec) { // [B*N]
  int tok = blockIdx.x;
  int t   = threadIdx.x;
  int id  = ids[tok];
  const float* te = text_table + (size_t)id * E_;
  int e0 = 2 * t, e1 = 2 * t + 1;
  X[(size_t)tok * 512 + t] = (u32)f2bf(te[e0]) | ((u32)f2bf(te[e1]) << 16);

  float p0 = npos[tok * 4 + 0], p1 = npos[tok * 4 + 1];
  float p2 = npos[tok * 4 + 2], p3 = npos[tok * 4 + 3];
  float a = posb[e0] + p0 * posW[e0] + p1 * posW[512 + e0]
                     + p2 * posW[1024 + e0] + p3 * posW[1536 + e0];
  float b = posb[e1] + p0 * posW[e1] + p1 * posW[512 + e1]
                     + p2 * posW[1024 + e1] + p3 * posW[1536 + e1];
  X[(size_t)tok * 512 + 256 + t] = (u32)f2bf(a) | ((u32)f2bf(b) << 16);

  if (t == 0) mvec[tok] = (id == 0) ? 1.0f : 0.0f;
}

// Gather the two query tokens per batch into a dense [B*2, 1024] bf16 matrix.
__global__ void __launch_bounds__(256)
gather_xq(const u32* __restrict__ X, u32* __restrict__ XQ) {
  int r   = blockIdx.x;                 // b*2 + i
  int tok = (r >> 1) * N_ + (r & 1);
  int t   = threadIdx.x;
  XQ[(size_t)r * 512 + t]       = X[(size_t)tok * 512 + t];
  XQ[(size_t)r * 512 + 256 + t] = X[(size_t)tok * 512 + 256 + t];
}

// ---------------------------------------------------------------------------
// bf16 GEMM: C[M,N] = A[M,K] @ W + bias.  A row-major bf16, W pre-packed.
// Block = 256 threads = 8 waves; each wave computes a 32x64 strip:
// 2 A-fragments x 4 B-fragments = 8 v_wmma_f32_16x16x32_bf16 per K-step.
// Two named fragment sets give software double-buffering (loads of k+1 in
// flight while WMMAs of k execute).  grid = (M/32, N/512).
// __launch_bounds__(256, 1) lifts the per-wave VGPR cap so the 64 acc +
// 96 fragment VGPRs live entirely in registers (no scratch spills).
// Requires: M % 32 == 0, N % 512 == 0, (K/32) even.  True for all uses.
// ---------------------------------------------------------------------------
struct FragSet { Frag a0, a1, b0, b1, b2, b3; };

__global__ void __launch_bounds__(256, 1)
gemm_bf16(const u16* __restrict__ A,
          const u32* __restrict__ Wpk,
          const float* __restrict__ bias,
          u16* __restrict__ Cb,     // bf16 out (or null)
          float* __restrict__ Cf,   // f32 out (or null)
          int M, int N, int K) {
  const int lane   = threadIdx.x & 31;
  const int wave   = threadIdx.x >> 5;
  const int mt     = blockIdx.x;                 // 32-row strip
  const int nt0    = blockIdx.y * 32 + wave * 4; // first of 4 n-tiles
  const int ktiles = K >> 5;
  const int ntiles = N >> 4;
  const int k0     = (lane >> 4) << 3;           // 0 or 8 bf16 elems

  const u16* Arow0 = A + (size_t)(mt * 32 + (lane & 15)) * K;
  const u16* Arow1 = Arow0 + (size_t)16 * K;
  const u32* Wb    = Wpk + (size_t)nt0 * 256 + lane * 8;

  v8f acc[8];
#pragma unroll
  for (int j = 0; j < 8; ++j)
#pragma unroll
    for (int e = 0; e < 8; ++e) acc[j][e] = 0.0f;

  auto loadFrags = [&](FragSet& f, int kt) {
    const u16* ap0 = Arow0 + (kt << 5) + k0;
    const u16* ap1 = Arow1 + (kt << 5) + k0;
    // 16-bit A 16x32 layout: lane holds K[k0..k0+7] and K[k0+16..k0+23]
    f.a0.q[0] = *(const u32x4*)(ap0);
    f.a0.q[1] = *(const u32x4*)(ap0 + 16);
    f.a1.q[0] = *(const u32x4*)(ap1);
    f.a1.q[1] = *(const u32x4*)(ap1 + 16);
    const u32* wb = Wb + (size_t)kt * (ntiles << 8);
    f.b0.q[0] = *(const u32x4*)(wb);        f.b0.q[1] = *(const u32x4*)(wb + 4);
    f.b1.q[0] = *(const u32x4*)(wb + 256);  f.b1.q[1] = *(const u32x4*)(wb + 260);
    f.b2.q[0] = *(const u32x4*)(wb + 512);  f.b2.q[1] = *(const u32x4*)(wb + 516);
    f.b3.q[0] = *(const u32x4*)(wb + 768);  f.b3.q[1] = *(const u32x4*)(wb + 772);
  };
  auto mm8 = [&](const FragSet& f) {
    acc[0] = __builtin_amdgcn_wmma_f32_16x16x32_bf16(false, f.a0.v, false, f.b0.v, (short)0, acc[0], false, false);
    acc[1] = __builtin_amdgcn_wmma_f32_16x16x32_bf16(false, f.a0.v, false, f.b1.v, (short)0, acc[1], false, false);
    acc[2] = __builtin_amdgcn_wmma_f32_16x16x32_bf16(false, f.a0.v, false, f.b2.v, (short)0, acc[2], false, false);
    acc[3] = __builtin_amdgcn_wmma_f32_16x16x32_bf16(false, f.a0.v, false, f.b3.v, (short)0, acc[3], false, false);
    acc[4] = __builtin_amdgcn_wmma_f32_16x16x32_bf16(false, f.a1.v, false, f.b0.v, (short)0, acc[4], false, false);
    acc[5] = __builtin_amdgcn_wmma_f32_16x16x32_bf16(false, f.a1.v, false, f.b1.v, (short)0, acc[5], false, false);
    acc[6] = __builtin_amdgcn_wmma_f32_16x16x32_bf16(false, f.a1.v, false, f.b2.v, (short)0, acc[6], false, false);
    acc[7] = __builtin_amdgcn_wmma_f32_16x16x32_bf16(false, f.a1.v, false, f.b3.v, (short)0, acc[7], false, false);
  };

  FragSet f0, f1;
  loadFrags(f0, 0);
  int kt = 0;
  for (; kt + 2 <= ktiles; kt += 2) {
    loadFrags(f1, kt + 1);                     // loads in flight over mm8(f0)
    if (kt + 2 < ktiles)
      __builtin_prefetch(Arow0 + ((kt + 2) << 5) + k0, 0, 1);  // global_prefetch_b8
    mm8(f0);
    if (kt + 2 < ktiles) loadFrags(f0, kt + 2);
    mm8(f1);
  }
  if (kt < ktiles) mm8(f0);                    // never taken: ktiles is even

  // C/D layout: VGPR e -> M = e + 8*(lane>=16), N = lane%16
  const int mbase = mt * 32 + ((lane >> 4) << 3);
#pragma unroll
  for (int half = 0; half < 2; ++half) {
#pragma unroll
    for (int j = 0; j < 4; ++j) {
      const int n  = (nt0 + j) * 16 + (lane & 15);
      const float bv = bias[n];
#pragma unroll
      for (int e = 0; e < 8; ++e) {
        const float val = acc[half * 4 + j][e] + bv;
        const size_t idx = (size_t)(mbase + half * 16 + e) * N + n;
        if (Cb) Cb[idx] = f2bf(val);
        if (Cf) Cf[idx] = val;
      }
    }
  }
}

// ---------------------------------------------------------------------------
// Attention for the 2 needed query tokens per (b,h): scores, outer-product
// padding mask, exact softmax over 100 keys, weighted sum of V.
// Block = 128 threads, grid = B*H.
// ---------------------------------------------------------------------------
__global__ void __launch_bounds__(128)
attention(const u16* __restrict__ Q,   // [B*2, 1024] bf16
          const u16* __restrict__ Kk,  // [B*N, 1024] bf16
          const u16* __restrict__ Vv,  // [B*N, 1024] bf16
          const float* __restrict__ mvec,
          u16* __restrict__ Att) {     // [B*2, 1024] bf16
  __shared__ float lq[2][DEPTH_];
  __shared__ float sc[2][128];
  __shared__ float red[128];
  int b = blockIdx.x >> 3;
  int h = blockIdx.x & 7;
  int t = threadIdx.x;

  lq[0][t] = bf2f(Q[(size_t)(b * 2 + 0) * DM_ + h * DEPTH_ + t]);
  lq[1][t] = bf2f(Q[(size_t)(b * 2 + 1) * DM_ + h * DEPTH_ + t]);
  __syncthreads();

  float m0 = mvec[b * N_ + 0], m1 = mvec[b * N_ + 1];
  const float scale = 0.08838834764831845f;   // 1/sqrt(128)
  if (t < N_) {
    const u16* kr = Kk + (size_t)(b * N_ + t) * DM_ + h * DEPTH_;
    float s0 = 0.f, s1 = 0.f;
    for (int d = 0; d < DEPTH_; ++d) {
      float kv = bf2f(kr[d]);
      s0 += lq[0][d] * kv;
      s1 += lq[1][d] * kv;
    }
    float mk = mvec[b * N_ + t];
    sc[0][t] = s0 * scale + m0 * mk * NEG_INF_F;
    sc[1][t] = s1 * scale + m1 * mk * NEG_INF_F;
  } else {
    sc[0][t] = -1e30f;
    sc[1][t] = -1e30f;
  }
  __syncthreads();

  for (int i = 0; i < 2; ++i) {
    red[t] = sc[i][t];
    __syncthreads();
    for (int s = 64; s > 0; s >>= 1) {
      if (t < s) red[t] = fmaxf(red[t], red[t + s]);
      __syncthreads();
    }
    float mx = red[0];
    __syncthreads();
    float p = __expf(sc[i][t] - mx);   // padding lanes underflow to 0
    red[t] = p;
    __syncthreads();
    for (int s = 64; s > 0; s >>= 1) {
      if (t < s) red[t] += red[t + s];
      __syncthreads();
    }
    float sum = red[0];
    __syncthreads();
    sc[i][t] = p / sum;
    __syncthreads();
  }

  float a0 = 0.f, a1 = 0.f;
  for (int n = 0; n < N_; ++n) {
    float vv = bf2f(Vv[(size_t)(b * N_ + n) * DM_ + h * DEPTH_ + t]);
    a0 += sc[0][n] * vv;
    a1 += sc[1][n] * vv;
  }
  Att[(size_t)(b * 2 + 0) * DM_ + h * DEPTH_ + t] = f2bf(a0);
  Att[(size_t)(b * 2 + 1) * DM_ + h * DEPTH_ + t] = f2bf(a1);
}

// cand_emb = cand_pos @ cand_W + cand_b  -> cat[:, 0:512] (bf16)
__global__ void __launch_bounds__(256)
cand_cat(const float* __restrict__ cand_pos,
         const float* __restrict__ candW,
         const float* __restrict__ candb,
         u16* __restrict__ cat) {   // [B, 1536] bf16
  int b = blockIdx.x, t = threadIdx.x;
  float p0 = cand_pos[b * 4 + 0], p1 = cand_pos[b * 4 + 1];
  float p2 = cand_pos[b * 4 + 2], p3 = cand_pos[b * 4 + 3];
  for (int e = t; e < E_; e += 256) {
    float v = candb[e] + p0 * candW[e] + p1 * candW[512 + e]
                       + p2 * candW[1024 + e] + p3 * candW[1536 + e];
    cat[(size_t)b * 1536 + e] = f2bf(v);
  }
}

// pooled = max(attWo[:,0,:], attWo[:,1,:]) -> cat[:, 512:1536] (bf16)
__global__ void __launch_bounds__(256)
pool_cat(const float* __restrict__ awo,   // [B*2, 1024] f32
         u16* __restrict__ cat) {
  int b = blockIdx.x, t = threadIdx.x;
  for (int j = t; j < DM_; j += 256) {
    float v = fmaxf(awo[(size_t)(2 * b) * DM_ + j],
                    awo[(size_t)(2 * b + 1) * DM_ + j]);
    cat[(size_t)b * 1536 + 512 + j] = f2bf(v);
  }
}

// out[b] = -sum(l2norm(enc)*l2norm(field_emb))   (with max(.,1e-12) clamps)
__global__ void __launch_bounds__(256)
cosine_loss(const float* __restrict__ enc,   // [B,512] f32
            const float* __restrict__ ftab,  // [64,512]
            const int* __restrict__ fid,
            float* __restrict__ out) {
  __shared__ float saa[256], sbb[256], sab[256];
  int b = blockIdx.x, t = threadIdx.x;
  const float* a = enc + (size_t)b * E_;
  const float* f = ftab + (size_t)fid[b] * E_;
  float aa = 0.f, bb = 0.f, ab = 0.f;
  for (int e = t; e < E_; e += 256) {
    float av = a[e], fv = f[e];
    aa += av * av; bb += fv * fv; ab += av * fv;
  }
  saa[t] = aa; sbb[t] = bb; sab[t] = ab;
  __syncthreads();
  for (int s = 128; s > 0; s >>= 1) {
    if (t < s) { saa[t] += saa[t + s]; sbb[t] += sbb[t + s]; sab[t] += sab[t + s]; }
    __syncthreads();
  }
  if (t == 0)
    out[b] = -sab[0] * rsqrtf(fmaxf(saa[0], 1e-12f))
                     * rsqrtf(fmaxf(sbb[0], 1e-12f));
}

// ---------------------------------------------------------------------------
extern "C" void kernel_launch(void* const* d_in, const int* in_sizes, int n_in,
                              void* d_out, int out_size, void* d_ws, size_t ws_size,
                              hipStream_t stream) {
  (void)in_sizes; (void)n_in; (void)out_size; (void)ws_size;

  const int*   field_id  = (const int*)d_in[0];
  const float* cand_pos  = (const float*)d_in[1];
  const int*   ntext     = (const int*)d_in[2];
  const float* npos      = (const float*)d_in[3];
  const float* text_tab  = (const float*)d_in[4];
  const float* field_tab = (const float*)d_in[5];
  const float* cand_W    = (const float*)d_in[6];
  const float* cand_b    = (const float*)d_in[7];
  const float* pos_W     = (const float*)d_in[8];
  const float* pos_b     = (const float*)d_in[9];
  const float* Wq = (const float*)d_in[10]; const float* bq = (const float*)d_in[11];
  const float* Wk = (const float*)d_in[12]; const float* bk = (const float*)d_in[13];
  const float* Wv = (const float*)d_in[14]; const float* bv = (const float*)d_in[15];
  const float* Wo = (const float*)d_in[16]; const float* bo = (const float*)d_in[17];
  const float* pW = (const float*)d_in[18]; const float* pb = (const float*)d_in[19];
  float* out = (float*)d_out;

  char* ws = (char*)d_ws;
  size_t off = 0;
  auto alloc = [&](size_t bytes) -> char* {
    char* p = ws + off;
    off = (off + bytes + 255) & ~(size_t)255;
    return p;
  };

  const size_t TOK = (size_t)B_ * N_;                 // 51200
  u32* wqpk = (u32*)alloc((size_t)DM_ * DM_ * 2);     // packed bf16 weights
  u32* wkpk = (u32*)alloc((size_t)DM_ * DM_ * 2);
  u32* wvpk = (u32*)alloc((size_t)DM_ * DM_ * 2);
  u32* wopk = (u32*)alloc((size_t)DM_ * DM_ * 2);
  u32* pppk = (u32*)alloc((size_t)1536 * E_ * 2);
  u32* xb   = (u32*)alloc(TOK * DM_ * 2);             // x bf16
  u32* xqb  = (u32*)alloc((size_t)B_ * 2 * DM_ * 2);  // gathered query rows
  u16* qb   = (u16*)alloc((size_t)B_ * 2 * DM_ * 2);
  u16* kb   = (u16*)alloc(TOK * DM_ * 2);
  u16* vb   = (u16*)alloc(TOK * DM_ * 2);
  u16* attb = (u16*)alloc((size_t)B_ * 2 * DM_ * 2);
  float* awo  = (float*)alloc((size_t)B_ * 2 * DM_ * 4);
  u16* catb = (u16*)alloc((size_t)B_ * 1536 * 2);
  float* encf = (float*)alloc((size_t)B_ * E_ * 4);
  float* mvec = (float*)alloc(TOK * 4);

  // 1) pack weights into WMMA B-fragment layout
  pack_w<<<(DM_ * DM_ / 512), 256, 0, stream>>>(Wq, wqpk, DM_, DM_);
  pack_w<<<(DM_ * DM_ / 512), 256, 0, stream>>>(Wk, wkpk, DM_, DM_);
  pack_w<<<(DM_ * DM_ / 512), 256, 0, stream>>>(Wv, wvpk, DM_, DM_);
  pack_w<<<(DM_ * DM_ / 512), 256, 0, stream>>>(Wo, wopk, DM_, DM_);
  pack_w<<<(1536 * E_ / 512), 256, 0, stream>>>(pW, pppk, 1536, E_);

  // 2) embeddings + pos projection -> x (bf16), padding mask
  build_x<<<B_ * N_, 256, 0, stream>>>(ntext, npos, text_tab, pos_W, pos_b, xb, mvec);

  // 3) gather the 2 query tokens per batch (only rows the pooling keeps)
  gather_xq<<<B_ * 2, 256, 0, stream>>>(xb, xqb);

  // 4) projections (WMMA bf16): Q over 1024 rows; K,V over all 51200 rows
  gemm_bf16<<<dim3(32, 2),   256, 0, stream>>>((const u16*)xqb, wqpk, bq, qb, nullptr, B_ * 2, DM_, DM_);
  gemm_bf16<<<dim3(1600, 2), 256, 0, stream>>>((const u16*)xb,  wkpk, bk, kb, nullptr, (int)TOK, DM_, DM_);
  gemm_bf16<<<dim3(1600, 2), 256, 0, stream>>>((const u16*)xb,  wvpk, bv, vb, nullptr, (int)TOK, DM_, DM_);

  // 5) masked softmax attention for the 2 query tokens
  attention<<<B_ * H_, 128, 0, stream>>>(qb, kb, vb, mvec, attb);

  // 6) output projection Wo (only 2 tokens/batch), f32 out for pooling
  gemm_bf16<<<dim3(32, 2), 256, 0, stream>>>(attb, wopk, bo, nullptr, awo, B_ * 2, DM_, DM_);

  // 7) concat(cand_emb, maxpool) -> proj GEMM -> cosine loss
  cand_cat<<<B_, 256, 0, stream>>>(cand_pos, cand_W, cand_b, catb);
  pool_cat<<<B_, 256, 0, stream>>>(awo, catb);
  gemm_bf16<<<dim3(16, 1), 256, 0, stream>>>(catb, pppk, pb, nullptr, encf, B_, E_, 1536);
  cosine_loss<<<B_, 256, 0, stream>>>(encf, field_tab, field_id, out);
}
